// SlidingWindowAttention_44281112822119
// MI455X (gfx1250) — compile-verified, matrix operand order checked by reading, not run on hardware
//
#include <hip/hip_runtime.h>

typedef __attribute__((ext_vector_type(16))) _Float16 v16h;
typedef __attribute__((ext_vector_type(8)))  _Float16 v8h;
typedef __attribute__((ext_vector_type(4)))  _Float16 v4h;
typedef __attribute__((ext_vector_type(8)))  float    v8f;
typedef __attribute__((ext_vector_type(4)))  float    v4f;

#define SEQ      4096
#define DMODEL   1024
#define NHEADS   16
#define HDIM     64
#define WINDOW   256

// ---------------------------------------------------------------------------
// A/B fragment loader for v_wmma_f32_16x16x32_f16, per ISA 7.12.2:
//   lanes 0-15 : row = lane,    elems[0:8)=K 0-7,  elems[8:16)=K 16-23
//   lanes 16-31: row = lane-16, elems[0:8)=K 8-15, elems[8:16)=K 24-31
// ---------------------------------------------------------------------------
__device__ __forceinline__ v16h load_frag16x32(const _Float16* __restrict__ base, int ld) {
  const int lane = threadIdx.x & 31;
  const int row  = lane & 15;
  const int koff = (lane >> 4) << 3;   // 0 or 8
  const _Float16* p = base + (size_t)row * ld + koff;
  v8h lo = *(const v8h*)(p);
  v8h hi = *(const v8h*)(p + 16);
  return __builtin_shufflevector(lo, hi, 0,1,2,3,4,5,6,7,8,9,10,11,12,13,14,15);
}

// ---------------------------------------------------------------------------
// 16-lane butterfly reductions on the VALU via DPP (no LDS pipeline use).
// quad_perm(1,0,3,2)=0xB1, quad_perm(2,3,0,1)=0x4E,
// row_half_mirror=0x141, row_mirror=0x140. All stay within a 16-lane row.
// ---------------------------------------------------------------------------
__device__ __forceinline__ float red16_max(float x) {
  int t;
  t = __builtin_amdgcn_update_dpp(0, __float_as_int(x), 0xB1,  0xf, 0xf, true);
  x = fmaxf(x, __int_as_float(t));
  t = __builtin_amdgcn_update_dpp(0, __float_as_int(x), 0x4E,  0xf, 0xf, true);
  x = fmaxf(x, __int_as_float(t));
  t = __builtin_amdgcn_update_dpp(0, __float_as_int(x), 0x141, 0xf, 0xf, true);
  x = fmaxf(x, __int_as_float(t));
  t = __builtin_amdgcn_update_dpp(0, __float_as_int(x), 0x140, 0xf, 0xf, true);
  x = fmaxf(x, __int_as_float(t));
  return x;
}

__device__ __forceinline__ float red16_sum(float x) {
  int t;
  t = __builtin_amdgcn_update_dpp(0, __float_as_int(x), 0xB1,  0xf, 0xf, true);
  x += __int_as_float(t);
  t = __builtin_amdgcn_update_dpp(0, __float_as_int(x), 0x4E,  0xf, 0xf, true);
  x += __int_as_float(t);
  t = __builtin_amdgcn_update_dpp(0, __float_as_int(x), 0x141, 0xf, 0xf, true);
  x += __int_as_float(t);
  t = __builtin_amdgcn_update_dpp(0, __float_as_int(x), 0x140, 0xf, 0xf, true);
  x += __int_as_float(t);
  return x;
}

// ---------------------------------------------------------------------------
// fp32 -> f16 elementwise convert (n multiple of 4)
// ---------------------------------------------------------------------------
__global__ void cvt_f32_f16(const float* __restrict__ in, _Float16* __restrict__ out, int n) {
  int i = (blockIdx.x * blockDim.x + threadIdx.x) * 4;
  if (i + 3 < n) {
    v4f x = *(const v4f*)(in + i);
    v4h o;
    o[0] = (_Float16)x[0]; o[1] = (_Float16)x[1];
    o[2] = (_Float16)x[2]; o[3] = (_Float16)x[3];
    *(v4h*)(out + i) = o;
  }
}

// ---------------------------------------------------------------------------
// C = A * B^T   (A: [M,K] f16 row-major, B: [N,K] f16 row-major, K=1024)
// Block = 8 waves = 128x64 output tile. The 64-row B panel is staged in LDS
// with global_load_async_to_lds_b128 (double buffered, ASYNCcnt tracked);
// all 8 waves read B fragments from LDS (8x traffic reduction on B).
// LDS row stride padded to 40 halves (80B) to avoid ds_load_b128 bank
// conflicts across the 16 fragment rows.
// MODE 0: f16 row-major [M,N]; MODE 1: f16 transposed [N,M]; MODE 2: f32.
// ---------------------------------------------------------------------------
#define BP_STRIDE 40                       // halves per LDS B-panel row
#define BP_BUF    (64 * BP_STRIDE)         // halves per buffer (2560)

template<int MODE>
__global__ __launch_bounds__(256)
void gemm_wmma_kernel(const _Float16* __restrict__ A,
                      const _Float16* __restrict__ B,
                      void* __restrict__ Cout,
                      int M, int N, int K) {
  extern __shared__ _Float16 smem[];       // 2 * BP_BUF halves = 10240 B

  const int mb = blockIdx.x >> 4;          // 32 m-blocks (128 rows each)
  const int np = blockIdx.x & 15;          // 16 n-panels (64 cols each)
  const int wv = threadIdx.x >> 5;         // 8 waves
  const int lane = threadIdx.x & 31;
  const int half = lane >> 4;
  const int col  = lane & 15;

  // async B staging: thread t moves one 16B chunk per K step
  const int brow = threadIdx.x >> 2;       // 0..63  (B panel row)
  const int bq   = threadIdx.x & 3;        // 0..3   (16B chunk within 32 halves)
  const _Float16* Bbase = B + (size_t)np * 64 * K;
  const _Float16* Abase = A + (size_t)(mb * 128 + wv * 16) * K;

  auto issue = [&](int k0, int buf) {
    const _Float16* src = Bbase + (size_t)brow * K + k0 + bq * 8;
    unsigned lds = (unsigned)(uintptr_t)(smem + buf * BP_BUF + brow * BP_STRIDE + bq * 8);
    unsigned long long ga = (unsigned long long)(uintptr_t)src;
    asm volatile("global_load_async_to_lds_b128 %0, %1, off"
                 :: "v"(lds), "v"(ga) : "memory");
  };

  v8f acc[4] = {};
  issue(0, 0);
  for (int k0 = 0, it = 0; k0 < K; k0 += 32, ++it) {
    const int buf = it & 1;
    if (k0 + 32 < K) {
      issue(k0 + 32, buf ^ 1);
      asm volatile("s_wait_asynccnt 0x1" ::: "memory");
    } else {
      asm volatile("s_wait_asynccnt 0x0" ::: "memory");
    }
    __syncthreads();

    v16h aF = load_frag16x32(Abase + k0, K);
    const _Float16* bp = smem + buf * BP_BUF;
#pragma unroll
    for (int j = 0; j < 4; ++j) {
      v16h bF = load_frag16x32(bp + j * 16 * BP_STRIDE, BP_STRIDE);
      acc[j] = __builtin_amdgcn_wmma_f32_16x16x32_f16(false, aF, false, bF,
                                                      (short)0, acc[j], false, false);
    }
    __syncthreads();                       // protect buf before re-issue
  }

#pragma unroll
  for (int j = 0; j < 4; ++j) {
#pragma unroll
    for (int r = 0; r < 8; ++r) {
      const int row = mb * 128 + wv * 16 + r + 8 * half;
      const int c   = np * 64 + j * 16 + col;
      const float v = acc[j][r];
      if (MODE == 0)      ((_Float16*)Cout)[(size_t)row * N + c] = (_Float16)v;
      else if (MODE == 1) ((_Float16*)Cout)[(size_t)c * M + row] = (_Float16)v;
      else                ((float*)Cout)[(size_t)row * N + c]    = v;
    }
  }
}

// ---------------------------------------------------------------------------
// Fused sliding-window flash attention.
//   Q, K : [SEQ, DMODEL] f16 (head slice = cols h*64 .. h*64+63)
//   Vt   : [DMODEL, SEQ] f16 (row = h*64+d, contiguous along j)
//   Aout : [SEQ, DMODEL] f16
// One wave per (head, 16-query tile); 32 keys per iteration.
// Softmax runs in the exp2 domain: Q is pre-scaled by 0.125*log2(e), so each
// probability costs v_sub + v_exp_f32. Key blocks fully inside the window
// ("full" blocks, a wave-uniform SALU test) skip all mask compares/selects;
// only the <=3 edge blocks per tile take the masked path.
// ---------------------------------------------------------------------------
__global__ __launch_bounds__(256)
void attn_kernel(const _Float16* __restrict__ Q,
                 const _Float16* __restrict__ Km,
                 const _Float16* __restrict__ Vt,
                 _Float16* __restrict__ Aout) {
  __shared__ _Float16 lds_p[8][16 * 32];

  const int wv   = threadIdx.x >> 5;
  const int wid  = blockIdx.x * (blockDim.x >> 5) + wv;
  const int h    = wid >> 8;
  const int qt   = wid & 255;
  const int i0   = qt * 16;
  const int lane = threadIdx.x & 31;
  const int half = lane >> 4;
  const int col  = lane & 15;

  // Q fragments with 1/sqrt(64) * log2(e) pre-folded
  const _Float16* qbase = Q + (size_t)i0 * DMODEL + h * HDIM;
  v16h qF0 = load_frag16x32(qbase,      DMODEL);
  v16h qF1 = load_frag16x32(qbase + 32, DMODEL);
  const _Float16 qscale = (_Float16)(0.125f * 1.44269504089f);
#pragma unroll
  for (int e = 0; e < 16; ++e) {
    qF0[e] = qF0[e] * qscale;
    qF1[e] = qF1[e] * qscale;
  }

  v8f   acc[4] = {};
  float mrow[8], lrow[8];
#pragma unroll
  for (int r = 0; r < 8; ++r) { mrow[r] = -1e30f; lrow[r] = 0.f; }

  int jlo = i0 - (WINDOW - 1); if (jlo < 0) jlo = 0;
  const int jb0 = jlo & ~31;
  const int jb1 = i0 + 15;

  for (int jb = jb0; jb <= jb1; jb += 32) {
    // ---- scores: two 16-key subtiles, contraction over d (two 32-halves)
    v8f sc[2];
#pragma unroll
    for (int s = 0; s < 2; ++s) {
      const _Float16* kb = Km + (size_t)(jb + 16 * s) * DMODEL + h * HDIM;
      v16h kF0 = load_frag16x32(kb,      DMODEL);
      v16h kF1 = load_frag16x32(kb + 32, DMODEL);
      v8f z = {};
      z = __builtin_amdgcn_wmma_f32_16x16x32_f16(false, qF0, false, kF0, (short)0, z, false, false);
      z = __builtin_amdgcn_wmma_f32_16x16x32_f16(false, qF1, false, kF1, (short)0, z, false, false);
      sc[s] = z;
    }

    // block fully inside every row's window? (wave-uniform SALU test)
    const bool full = (jb + 31 <= i0) && (jb >= i0 - (WINDOW - 16));

    float p0v[8], p1v[8];
    if (full) {
      // ---- unmasked fast path: no compares, no selects
#pragma unroll
      for (int r = 0; r < 8; ++r) {
        const float v0 = sc[0][r];
        const float v1 = sc[1][r];
        float m = red16_max(fmaxf(v0, v1));
        m = fmaxf(m, mrow[r]);
        const float cf = __builtin_amdgcn_exp2f(mrow[r] - m);
        mrow[r] = m;
        const float p0 = __builtin_amdgcn_exp2f(v0 - m);
        const float p1 = __builtin_amdgcn_exp2f(v1 - m);
        const float ls = red16_sum(p0 + p1);
        lrow[r] = lrow[r] * cf + ls;
#pragma unroll
        for (int t = 0; t < 4; ++t) acc[t][r] *= cf;
        p0v[r] = p0; p1v[r] = p1;
      }
    } else {
      // ---- masked edge path: single unsigned-range compare per element
#pragma unroll
      for (int r = 0; r < 8; ++r) {
        const int row = i0 + r + 8 * half;
        const int dj0 = row - (jb + col);        // valid iff 0 <= dj < WINDOW
        const int dj1 = row - (jb + 16 + col);
        const float v0 = ((unsigned)dj0 < WINDOW) ? sc[0][r] : -1e30f;
        const float v1 = ((unsigned)dj1 < WINDOW) ? sc[1][r] : -1e30f;

        float m = red16_max(fmaxf(v0, v1));
        m = fmaxf(m, mrow[r]);
        const float cf = __builtin_amdgcn_exp2f(mrow[r] - m);
        mrow[r] = m;

        const float p0 = (v0 > -1e29f) ? __builtin_amdgcn_exp2f(v0 - m) : 0.f;
        const float p1 = (v1 > -1e29f) ? __builtin_amdgcn_exp2f(v1 - m) : 0.f;
        const float ls = red16_sum(p0 + p1);
        lrow[r] = lrow[r] * cf + ls;
#pragma unroll
        for (int t = 0; t < 4; ++t) acc[t][r] *= cf;
        p0v[r] = p0; p1v[r] = p1;
      }
    }

    // ---- P: C-layout f32 -> LDS -> A-layout f16 fragment
#pragma unroll
    for (int r = 0; r < 8; ++r) {
      lds_p[wv][(r + 8 * half) * 32 + col]      = (_Float16)p0v[r];
      lds_p[wv][(r + 8 * half) * 32 + 16 + col] = (_Float16)p1v[r];
    }
    asm volatile("s_wait_dscnt 0" ::: "memory");
    v16h pF = load_frag16x32(&lds_p[wv][0], 32);

    // ---- O += P * V   (B = Vt rows, contiguous along j)
#pragma unroll
    for (int t = 0; t < 4; ++t) {
      v16h vF = load_frag16x32(Vt + (size_t)(h * HDIM + t * 16) * SEQ + jb, SEQ);
      acc[t] = __builtin_amdgcn_wmma_f32_16x16x32_f16(false, pF, false, vF,
                                                      (short)0, acc[t], false, false);
    }
  }

  // ---- normalize and store
#pragma unroll
  for (int r = 0; r < 8; ++r) {
    const float inv = 1.f / lrow[r];
    const int   row = i0 + r + 8 * half;
#pragma unroll
    for (int t = 0; t < 4; ++t) {
      Aout[(size_t)row * DMODEL + h * HDIM + t * 16 + col] = (_Float16)(acc[t][r] * inv);
    }
  }
}

// ---------------------------------------------------------------------------
// Host launcher
// ---------------------------------------------------------------------------
extern "C" void kernel_launch(void* const* d_in, const int* in_sizes, int n_in,
                              void* d_out, int out_size, void* d_ws, size_t ws_size,
                              hipStream_t stream) {
  const float* x  = (const float*)d_in[0];
  const float* Wq = (const float*)d_in[1];
  const float* Wk = (const float*)d_in[2];
  const float* Wv = (const float*)d_in[3];
  const float* Wo = (const float*)d_in[4];

  const size_t NX = (size_t)SEQ * DMODEL;      // 4M elems
  const size_t NW = (size_t)DMODEL * DMODEL;   // 1M elems

  _Float16* ws  = (_Float16*)d_ws;
  _Float16* xh  = ws;                // [SEQ, DMODEL]
  _Float16* wqh = xh  + NX;          // [DMODEL, DMODEL]
  _Float16* wkh = wqh + NW;
  _Float16* wvh = wkh + NW;
  _Float16* woh = wvh + NW;
  _Float16* qh  = woh + NW;          // [SEQ, DMODEL]
  _Float16* kh  = qh  + NX;          // [SEQ, DMODEL]
  _Float16* vt  = kh  + NX;          // [DMODEL, SEQ]  (transposed V)
  _Float16* ah  = vt  + NX;          // [SEQ, DMODEL]

  // 1) converts
  cvt_f32_f16<<<(int)(NX / 4 / 256), 256, 0, stream>>>(x,  xh,  (int)NX);
  cvt_f32_f16<<<(int)(NW / 4 / 256), 256, 0, stream>>>(Wq, wqh, (int)NW);
  cvt_f32_f16<<<(int)(NW / 4 / 256), 256, 0, stream>>>(Wk, wkh, (int)NW);
  cvt_f32_f16<<<(int)(NW / 4 / 256), 256, 0, stream>>>(Wv, wvh, (int)NW);
  cvt_f32_f16<<<(int)(NW / 4 / 256), 256, 0, stream>>>(Wo, woh, (int)NW);

  // 2) QKV projections: 32 m-blocks x 16 n-panels, 8 waves/block,
  //    10240B dynamic LDS for the double-buffered async B panel
  const int blocks = 32 * 16;                    // 512
  const size_t shmem = 2 * BP_BUF * sizeof(_Float16);
  gemm_wmma_kernel<0><<<blocks, 256, shmem, stream>>>(xh, wqh, (void*)qh, SEQ, DMODEL, DMODEL);
  gemm_wmma_kernel<0><<<blocks, 256, shmem, stream>>>(xh, wkh, (void*)kh, SEQ, DMODEL, DMODEL);
  gemm_wmma_kernel<1><<<blocks, 256, shmem, stream>>>(xh, wvh, (void*)vt, SEQ, DMODEL, DMODEL);

  // 3) fused sliding-window attention: 16 heads * 256 qtiles = 4096 waves
  attn_kernel<<<512, 256, 0, stream>>>(qh, kh, vt, ah);

  // 4) output projection -> fp32
  gemm_wmma_kernel<2><<<blocks, 256, shmem, stream>>>(ah, woh, d_out, SEQ, DMODEL, DMODEL);
}
